// ABMIL_39496519254706
// MI455X (gfx1250) — matrix operand order repaired and checked by reading, not compiled, not run on hardware
//
#include <hip/hip_runtime.h>

// ---------------------------------------------------------------------------
// ABMIL fused forward for MI455X (gfx1250, wave32, WMMA).
//
// Roofline: x read = 512MB -> 22us @ 23.3TB/s.  86 GFLOP in bf16 WMMA
// (16x16x32) ~ matches memory time; fp32 WMMA (16x16x4) would be ~8x slower.
// So: convert to bf16 in-register, accumulate fp32 in v_wmma_f32_16x16x32_bf16.
// ---------------------------------------------------------------------------

#define FDIM 1024
#define HDIM 256
#define DDIM 128
#define NTOK 16384
#define BATCH 8
#define MT 128   // tokens per workgroup

typedef __attribute__((ext_vector_type(16))) __bf16 v16bf;
typedef __attribute__((ext_vector_type(8)))  float  v8f;

union BF16Frag {
    v16bf v;
    uint4 q[2];
};

// fp32 -> bf16, round-to-nearest-even
__device__ __forceinline__ unsigned short f2bf_bits(float f) {
    union { float f; unsigned u; } c; c.f = f;
    unsigned r = c.u + 0x7FFFu + ((c.u >> 16) & 1u);
    return (unsigned short)(r >> 16);
}
__device__ __forceinline__ unsigned pack2(float lo, float hi) {
    return (unsigned)f2bf_bits(lo) | ((unsigned)f2bf_bits(hi) << 16);
}

// ---------------------------------------------------------------------------
// Pass 0: convert + transpose weights to bf16 (tiny, done once per launch)
//   wt1[n][k] = bf16(W1[k][n])   (256 x 1024)
//   wvt[d][h] = bf16(Wv[h][d])   (128 x 256)
//   wut[d][h] = bf16(Wu[h][d])   (128 x 256)
// ---------------------------------------------------------------------------
__global__ __launch_bounds__(256) void abmil_cvt(
    const float* __restrict__ W1, const float* __restrict__ Wv,
    const float* __restrict__ Wu,
    unsigned short* __restrict__ wt1, unsigned short* __restrict__ wvt,
    unsigned short* __restrict__ wut)
{
    int idx = blockIdx.x * 256 + threadIdx.x;
    if (idx < FDIM * HDIM) {
        int k = idx >> 8, n = idx & 255;
        wt1[n * FDIM + k] = f2bf_bits(W1[idx]);
    } else if (idx < FDIM * HDIM + HDIM * DDIM) {
        int j = idx - FDIM * HDIM;
        int hh = j >> 7, d = j & 127;
        wvt[d * HDIM + hh] = f2bf_bits(Wv[j]);
    } else if (idx < FDIM * HDIM + 2 * HDIM * DDIM) {
        int j = idx - FDIM * HDIM - HDIM * DDIM;
        int hh = j >> 7, d = j & 127;
        wut[d * HDIM + hh] = f2bf_bits(Wu[j]);
    }
}

// ---------------------------------------------------------------------------
// Pass 1: fused  h = GELU(x@W1+b1);  logit = (tanh(h@Wv+bv)*sigm(h@Wu+bu))@Ww+bw
// Grid: 1024 WGs (8 batches * 128 token-blocks), 256 threads = 8 waves.
// Wave w owns tokens [w*16, w*16+16) of its 128-token block, computes the
// full 16x256 h tile (16 WMMA n-tiles) and then its 16 attention logits.
// ---------------------------------------------------------------------------
__global__ __launch_bounds__(256) void abmil_fused(
    const float* __restrict__ x,  const float* __restrict__ b1,
    const float* __restrict__ bv, const float* __restrict__ bu,
    const float* __restrict__ Ww, const float* __restrict__ bw,
    const unsigned short* __restrict__ wt1,
    const unsigned short* __restrict__ wvt,
    const unsigned short* __restrict__ wut,
    float* __restrict__ h_out, float* __restrict__ a_out)
{
    // 30720 B of LDS, two aliased phase layouts (pitch 40 halfs = 80B, 16B-aligned rows)
    __shared__ __align__(16) unsigned short smem[15360];
    unsigned short* xs  = smem;               // phase1: x tile   [128][40] bf16
    unsigned short* w1s = smem + 128 * 40;    // phase1: W1 chunk [256][40] bf16 (row=n)
    unsigned short* wvs = smem;               // phase2: Wv chunk [128][40]
    unsigned short* wus = smem + 128 * 40;    // phase2: Wu chunk [128][40]
    unsigned short* hst = smem + 256 * 40;    // phase2: h stage  [8][16][40]

    const int tid  = threadIdx.x;
    const int wave = tid >> 5;
    const int lane = tid & 31;
    const int ln   = lane & 15;   // A: row M; B: col N; C: col N
    const int hi   = lane >> 4;   // K-half selector / C row-half selector
    const int blk  = blockIdx.x;
    const int b    = blk >> 7;                 // 128 blocks per batch
    const int tokBase = (blk & 127) * MT;

    v8f c[16];
    #pragma unroll
    for (int i = 0; i < 16; ++i) c[i] = (v8f){0.f,0.f,0.f,0.f,0.f,0.f,0.f,0.f};

    // staging mapping for x: thread -> (row, 16-col half)
    const int sr = tid >> 1;
    const int sh = tid & 1;
    const float* xsrc = x + ((size_t)b * NTOK + tokBase + sr) * FDIM + sh * 16;

    // ---------------- GEMM1: h = x @ W1 ----------------
    for (int ks = 0; ks < 32; ++ks) {
        const int kb = ks * 32;
        // stage x chunk (fp32 -> bf16) into LDS
        {
            const float4* xp = (const float4*)(xsrc + kb);
            float4 f0 = xp[0], f1 = xp[1], f2 = xp[2], f3 = xp[3];
            if (ks < 31) __builtin_prefetch(xsrc + kb + 32, 0, 1); // global_prefetch_b8
            uint4 p0, p1;
            p0.x = pack2(f0.x, f0.y); p0.y = pack2(f0.z, f0.w);
            p0.z = pack2(f1.x, f1.y); p0.w = pack2(f1.z, f1.w);
            p1.x = pack2(f2.x, f2.y); p1.y = pack2(f2.z, f2.w);
            p1.z = pack2(f3.x, f3.y); p1.w = pack2(f3.z, f3.w);
            *(uint4*)&xs[sr * 40 + sh * 16]     = p0;
            *(uint4*)&xs[sr * 40 + sh * 16 + 8] = p1;
        }
        // stage W1 chunk (already bf16, pre-transposed): row n = tid, 64B copy
        {
            const uint4* wsrc = (const uint4*)(wt1 + tid * FDIM + kb);
            uint4 a0 = wsrc[0], a1 = wsrc[1], a2 = wsrc[2], a3 = wsrc[3];
            uint4* wdst = (uint4*)&w1s[tid * 40];
            wdst[0] = a0; wdst[1] = a1; wdst[2] = a2; wdst[3] = a3;
        }
        __syncthreads();

        // A fragment (16x32 bf16): lane(ln,hi) needs K run [hi*16,+16)
        BF16Frag af;
        const unsigned short* arow = &xs[(wave * 16 + ln) * 40];
        af.q[0] = *(const uint4*)(arow + hi * 16);
        af.q[1] = *(const uint4*)(arow + hi * 16 + 8);

        #pragma unroll
        for (int nt = 0; nt < 16; ++nt) {
            // B fragment (32x16): lane(ln,hi) needs col n=nt*16+ln, K run [hi*16,+16)
            BF16Frag bf;
            const unsigned short* brow = &w1s[(nt * 16 + ln) * 40];
            bf.q[0] = *(const uint4*)(brow + hi * 16);
            bf.q[1] = *(const uint4*)(brow + hi * 16 + 8);
            c[nt] = __builtin_amdgcn_wmma_f32_16x16x32_bf16(
                false, af.v, false, bf.v, (short)0, c[nt], false, false);
        }
        __syncthreads();
    }

    // ---------------- bias + exact GELU, write h ----------------
    #pragma unroll
    for (int nt = 0; nt < 16; ++nt) {
        const int col = nt * 16 + ln;                 // C layout: N = lane%16
        const float bb = b1[col];
        #pragma unroll
        for (int v = 0; v < 8; ++v) {                 // C layout: M = v + 8*hi
            float t = c[nt][v] + bb;
            float g = 0.5f * t * (1.0f + erff(t * 0.70710678118654752f));
            c[nt][v] = g;
            h_out[((size_t)b * NTOK + tokBase + wave * 16 + v + 8 * hi) * HDIM + col] = g;
        }
    }
    __syncthreads();   // LDS re-aliased below

    // ---------------- GEMM2/3: aV = h@Wv, aU = h@Wu ----------------
    v8f av[8], au[8];
    #pragma unroll
    for (int i = 0; i < 8; ++i) {
        av[i] = (v8f){0.f,0.f,0.f,0.f,0.f,0.f,0.f,0.f};
        au[i] = (v8f){0.f,0.f,0.f,0.f,0.f,0.f,0.f,0.f};
    }
    for (int k2 = 0; k2 < 8; ++k2) {
        const int kb2 = k2 * 32;
        // stage Wv/Wu chunks (shared by all waves)
        {
            const int r = tid >> 1, part = tid & 1;
            const uint4* vsrc = (const uint4*)(wvt + r * HDIM + kb2 + part * 16);
            const uint4* usrc = (const uint4*)(wut + r * HDIM + kb2 + part * 16);
            uint4 v0 = vsrc[0], v1 = vsrc[1], u0 = usrc[0], u1 = usrc[1];
            uint4* vd = (uint4*)&wvs[r * 40 + part * 16]; vd[0] = v0; vd[1] = v1;
            uint4* ud = (uint4*)&wus[r * 40 + part * 16]; ud[0] = u0; ud[1] = u1;
        }
        // stage this wave's h columns [kb2, kb2+32) as bf16 (C-layout -> row-major)
        {
            unsigned short* hrow = &hst[wave * 16 * 40];
            #pragma unroll
            for (int j = 0; j < 2; ++j) {
                const int nt = 2 * k2 + j;
                #pragma unroll
                for (int v = 0; v < 8; ++v)
                    hrow[(v + 8 * hi) * 40 + j * 16 + ln] = f2bf_bits(c[nt][v]);
            }
        }
        __syncthreads();

        BF16Frag af2;
        const unsigned short* ar = &hst[(wave * 16 + ln) * 40];
        af2.q[0] = *(const uint4*)(ar + hi * 16);
        af2.q[1] = *(const uint4*)(ar + hi * 16 + 8);

        #pragma unroll
        for (int nt2 = 0; nt2 < 8; ++nt2) {
            BF16Frag bfv, bfu;
            const unsigned short* vr = &wvs[(nt2 * 16 + ln) * 40];
            bfv.q[0] = *(const uint4*)(vr + hi * 16);
            bfv.q[1] = *(const uint4*)(vr + hi * 16 + 8);
            av[nt2] = __builtin_amdgcn_wmma_f32_16x16x32_bf16(
                false, af2.v, false, bfv.v, (short)0, av[nt2], false, false);
            const unsigned short* ur = &wus[(nt2 * 16 + ln) * 40];
            bfu.q[0] = *(const uint4*)(ur + hi * 16);
            bfu.q[1] = *(const uint4*)(ur + hi * 16 + 8);
            au[nt2] = __builtin_amdgcn_wmma_f32_16x16x32_bf16(
                false, af2.v, false, bfu.v, (short)0, au[nt2], false, false);
        }
        __syncthreads();
    }

    // ---------------- tanh * sigmoid * Ww, reduce over d ----------------
    float psum[8];
    #pragma unroll
    for (int v = 0; v < 8; ++v) psum[v] = 0.f;
    #pragma unroll
    for (int nt2 = 0; nt2 < 8; ++nt2) {
        const int d = nt2 * 16 + ln;
        const float bvv = bv[d], buv = bu[d], wwd = Ww[d];
        #pragma unroll
        for (int v = 0; v < 8; ++v) {
            float tv = tanhf(av[nt2][v] + bvv);
            float sv = 1.0f / (1.0f + expf(-(au[nt2][v] + buv)));
            psum[v] += tv * sv * wwd;
        }
    }
    const float bw0 = bw[0];
    #pragma unroll
    for (int v = 0; v < 8; ++v) {
        float s = psum[v];
        s += __shfl_xor(s, 1); s += __shfl_xor(s, 2);
        s += __shfl_xor(s, 4); s += __shfl_xor(s, 8);   // sum within 16-lane half
        if (ln == 0)
            a_out[(size_t)b * NTOK + tokBase + wave * 16 + v + 8 * hi] = s + bw0;
    }
}

// ---------------------------------------------------------------------------
// Pass 2: per-batch softmax over N; writes attn to out[2048..], zeroes slide slots
// ---------------------------------------------------------------------------
__global__ __launch_bounds__(256) void abmil_softmax(
    const float* __restrict__ a, const unsigned char* __restrict__ mask,
    float* __restrict__ out)
{
    const int b = blockIdx.x, t = threadIdx.x;
    __shared__ float red[256];
    const float* ab = a + (size_t)b * NTOK;
    const unsigned char* mb = mask + (size_t)b * NTOK;

    float mx = -3.0e38f;
    for (int i = t; i < NTOK; i += 256) {
        float v = mb[i] ? ab[i] : -3.0e38f;
        mx = fmaxf(mx, v);
    }
    red[t] = mx; __syncthreads();
    for (int s = 128; s > 0; s >>= 1) {
        if (t < s) red[t] = fmaxf(red[t], red[t + s]);
        __syncthreads();
    }
    mx = red[0]; __syncthreads();

    float sum = 0.f;
    float* attn = out + 2048 + (size_t)b * NTOK;
    for (int i = t; i < NTOK; i += 256) {
        float e = mb[i] ? expf(ab[i] - mx) : 0.f;
        sum += e;
        attn[i] = e;
    }
    red[t] = sum; __syncthreads();
    for (int s = 128; s > 0; s >>= 1) {
        if (t < s) red[t] += red[t + s];
        __syncthreads();
    }
    const float inv = 1.0f / red[0];
    for (int i = t; i < NTOK; i += 256) attn[i] *= inv;

    out[b * HDIM + t] = 0.f;   // zero slide-token slot for pass-3 atomics
}

// ---------------------------------------------------------------------------
// Pass 3: slide_tokens[b,:] = sum_n attn[b,n] * h[b,n,:]  (atomic fp32 adds)
// ---------------------------------------------------------------------------
__global__ __launch_bounds__(256) void abmil_pool(
    const float* __restrict__ h, float* __restrict__ out)
{
    const int b = blockIdx.y, chunk = blockIdx.x, t = threadIdx.x;
    __shared__ float at[256];
    const float* attn = out + 2048 + (size_t)b * NTOK + chunk * 256;
    at[t] = attn[t];
    __syncthreads();
    float acc = 0.f;
    const float* hb = h + ((size_t)b * NTOK + (size_t)chunk * 256) * HDIM + t;
    #pragma unroll 4
    for (int n = 0; n < 256; ++n) acc += at[n] * hb[(size_t)n * HDIM];
    atomicAdd(&out[b * HDIM + t], acc);
}

// ---------------------------------------------------------------------------
extern "C" void kernel_launch(void* const* d_in, const int* in_sizes, int n_in,
                              void* d_out, int out_size, void* d_ws, size_t ws_size,
                              hipStream_t stream) {
    const float*         x    = (const float*)d_in[0];
    const unsigned char* mask = (const unsigned char*)d_in[1];
    const float*         W1   = (const float*)d_in[2];
    const float*         b1   = (const float*)d_in[3];
    const float*         Wv   = (const float*)d_in[4];
    const float*         bv   = (const float*)d_in[5];
    const float*         Wu   = (const float*)d_in[6];
    const float*         bu   = (const float*)d_in[7];
    const float*         Ww   = (const float*)d_in[8];
    const float*         bw   = (const float*)d_in[9];
    float* out = (float*)d_out;

    char* w = (char*)d_ws;
    float*          h_ws = (float*)w;                                   // 128 MiB
    float*          a_ws = (float*)(w + (size_t)134217728);             // 512 KiB
    unsigned short* wt1  = (unsigned short*)(w + 134217728 + 524288);   // 512 KiB
    unsigned short* wvt  = (unsigned short*)(w + 134217728 + 1048576);  // 64 KiB
    unsigned short* wut  = (unsigned short*)(w + 134217728 + 1114112);  // 64 KiB

    abmil_cvt<<<1280, 256, 0, stream>>>(W1, Wv, Wu, wt1, wvt, wut);
    abmil_fused<<<1024, 256, 0, stream>>>(x, b1, bv, bu, Ww, bw,
                                          wt1, wvt, wut, h_ws, a_ws);
    abmil_softmax<<<8, 256, 0, stream>>>(a_ws, mask, out);
    abmil_pool<<<dim3(64, 8), 256, 0, stream>>>(h_ws, out);
}
